// GAT_57526791962991
// MI455X (gfx1250) — matrix operand order repaired and checked by reading, not compiled, not run on hardware
//
#include <hip/hip_runtime.h>

#define NN 20000
#define NE 100000
#define DIM 768
#define NEG_SLOPE 0.2f

typedef __attribute__((ext_vector_type(16))) __bf16 v16bf;
typedef __attribute__((ext_vector_type(8)))  __bf16 v8bf;
typedef __attribute__((ext_vector_type(8)))  float  v8f;

static __device__ __forceinline__ __bf16 f2bf(float f) {
  unsigned u = __float_as_uint(f);
  u += 0x7FFFu + ((u >> 16) & 1u);            // round-to-nearest-even
  unsigned short h = (unsigned short)(u >> 16);
  return __builtin_bit_cast(__bf16, h);
}

#define CAT16(lo, hi) __builtin_shufflevector(lo, hi, 0,1,2,3,4,5,6,7,8,9,10,11,12,13,14,15)
#define WMMA_BF16(a, b, c) \
  __builtin_amdgcn_wmma_f32_16x16x32_bf16(false, a, false, b, (short)0, c, false, false)

// float atomic-max via int-max (val>=0) / uint-min (val<0); dest initialized to -inf.
static __device__ __forceinline__ void atomicMaxF(float* addr, float val) {
  if (val >= 0.0f) atomicMax((int*)addr, __float_as_int(val));
  else             atomicMin((unsigned int*)addr, __float_as_uint(val));
}

// ---------------------------------------------------------------------------
// Pre-pass: fp32 -> bf16 conversions, hoisted out of the GEMM hot loop.
// ---------------------------------------------------------------------------
__global__ void cvt_x_bf16(const float* __restrict__ src, __bf16* __restrict__ dst, int n) {
  int i = blockIdx.x * 256 + threadIdx.x;
  if (i < n) dst[i] = f2bf(src[i]);
}

// W[k][n] (row-major) -> Wt[n][k] bf16 (K-contiguous per output column)
__global__ void cvt_w_t_bf16(const float* __restrict__ W, __bf16* __restrict__ Wt) {
  int i = blockIdx.x * 256 + threadIdx.x;     // over 768*768, writes coalesced
  if (i >= DIM * DIM) return;
  int n = i / DIM, k = i - n * DIM;
  Wt[i] = f2bf(W[(size_t)k * DIM + n]);
}

// ---------------------------------------------------------------------------
// GEMM: Z[20000,768] = X * W, bf16 WMMA, fp32 accumulate, NO LDS.
// Fragments loaded straight from L2-resident bf16 buffers as 16B b128 loads.
// Wave tile 32x32: 4 accumulators, 4 WMMAs/K-step, each fragment reused 2x.
// Block = 256 (8 wave32) -> 64x128 C tile. Grid (768/128, ceil(20000/64)).
//
// ISA 7.12.2 bf16 fragment layouts (wave32):
//  A 16x32: lanes 0-15 M=lane K{0..7,16..23}; lanes 16-31 M=lane-16 K{8..15,24..31}
//  B 32x16: lanes 0-15 N=lane K{0..7,16..23}; lanes 16-31 N=lane-16 K{8..15,24..31}
//  D:       VGPR v -> M = 8*half + v, N = lane&15
// ---------------------------------------------------------------------------
__global__ __launch_bounds__(256) void gat_gemm(const __bf16* __restrict__ Xb,
                                                const __bf16* __restrict__ Wt,
                                                float* __restrict__ Z) {
  const unsigned tid  = threadIdx.x;
  const unsigned lane = tid & 31u;
  const unsigned wave = tid >> 5;
  const unsigned m0   = (wave & 1u) * 32u;
  const unsigned n0   = (wave >> 1) * 32u;
  const unsigned gM   = blockIdx.y * 64u;
  const unsigned gN   = blockIdx.x * 128u;
  const unsigned half = lane >> 4;

  unsigned r0 = gM + m0 + (lane & 15u);       // A0 row, A1 row = +16
  unsigned r1 = r0 + 16u;
  if (r0 >= NN) r0 = NN - 1;                  // clamp loads; stores guarded below
  if (r1 >= NN) r1 = NN - 1;

  const __bf16* pa0 = Xb + (size_t)r0 * DIM + half * 8u;
  const __bf16* pa1 = Xb + (size_t)r1 * DIM + half * 8u;
  const __bf16* pb0 = Wt + (size_t)(gN + n0 + (lane & 15u)) * DIM + half * 8u;
  const __bf16* pb1 = pb0 + (size_t)16u * DIM;

  v8f c00 = {}, c01 = {}, c10 = {}, c11 = {};

#pragma unroll 2
  for (unsigned kt = 0; kt < DIM; kt += 32u) {
    v16bf a0 = CAT16(*(const v8bf*)(pa0 + kt), *(const v8bf*)(pa0 + kt + 16u));
    v16bf a1 = CAT16(*(const v8bf*)(pa1 + kt), *(const v8bf*)(pa1 + kt + 16u));
    v16bf b0 = CAT16(*(const v8bf*)(pb0 + kt), *(const v8bf*)(pb0 + kt + 16u));
    v16bf b1 = CAT16(*(const v8bf*)(pb1 + kt), *(const v8bf*)(pb1 + kt + 16u));
    c00 = WMMA_BF16(a0, b0, c00);
    c01 = WMMA_BF16(a0, b1, c01);
    c10 = WMMA_BF16(a1, b0, c10);
    c11 = WMMA_BF16(a1, b1, c11);
  }

  const unsigned rb0 = gM + m0 + half * 8u;
  const unsigned rb1 = rb0 + 16u;
  const unsigned cc  = gN + n0 + (lane & 15u);
#pragma unroll
  for (int v = 0; v < 8; ++v) {
    unsigned ra = rb0 + (unsigned)v, rb = rb1 + (unsigned)v;
    if (ra < NN) {
      Z[(size_t)ra * DIM + cc]       = c00[v];
      Z[(size_t)ra * DIM + cc + 16u] = c01[v];
    }
    if (rb < NN) {
      Z[(size_t)rb * DIM + cc]       = c10[v];
      Z[(size_t)rb * DIM + cc + 16u] = c11[v];
    }
  }
}

// ---------------------------------------------------------------------------
__global__ void gat_zero(float* __restrict__ p, int n) {
  int i = blockIdx.x * 256 + threadIdx.x;
  if (i < n) p[i] = 0.0f;
}

// per-node: el[n,h]=sum_d z*al, er likewise; also init m=-inf, denom=0
__global__ __launch_bounds__(256) void gat_attn(const float* __restrict__ Z,
                                                const float* __restrict__ al,
                                                const float* __restrict__ ar,
                                                float* __restrict__ el, float* __restrict__ er,
                                                float* __restrict__ m,  float* __restrict__ den,
                                                int H, int hshift) {
  __shared__ float sl[12], sr[12];
  const int n = blockIdx.x;
  if (threadIdx.x < (unsigned)H) { sl[threadIdx.x] = 0.0f; sr[threadIdx.x] = 0.0f; }
  __syncthreads();
  for (int i = threadIdx.x; i < DIM; i += 256) {
    float zv = Z[(size_t)n * DIM + i];
    int h = i >> hshift;
    atomicAdd(&sl[h], zv * al[i]);
    atomicAdd(&sr[h], zv * ar[i]);
  }
  __syncthreads();
  if (threadIdx.x < (unsigned)H) {
    el[n * H + threadIdx.x]  = sl[threadIdx.x];
    er[n * H + threadIdx.x]  = sr[threadIdx.x];
    m[n * H + threadIdx.x]   = __uint_as_float(0xFF800000u);  // -inf
    den[n * H + threadIdx.x] = 0.0f;
  }
}

__global__ void gat_edge_max(const int* __restrict__ src, const int* __restrict__ dst,
                             const float* __restrict__ el, const float* __restrict__ er,
                             float* __restrict__ elog, float* __restrict__ m, int H) {
  int i = blockIdx.x * 256 + threadIdx.x;
  if (i >= NE * H) return;
  int e = i / H, h = i - e * H;
  float v = el[src[e] * H + h] + er[dst[e] * H + h];
  v = (v >= 0.0f) ? v : NEG_SLOPE * v;
  elog[i] = v;
  atomicMaxF(&m[dst[e] * H + h], v);
}

__global__ void gat_edge_exp(const int* __restrict__ dst, float* __restrict__ elog,
                             const float* __restrict__ m, float* __restrict__ den, int H) {
  int i = blockIdx.x * 256 + threadIdx.x;
  if (i >= NE * H) return;
  int e = i / H, h = i - e * H;
  float ex = expf(elog[i] - m[dst[e] * H + h]);
  elog[i] = ex;                                   // reuse buffer: now holds exp(e-m)
  atomicAdd(&den[dst[e] * H + h], ex);
}

// one block per edge: acc[dst] += alpha[h] * z[src]; z & acc stay L2-resident
__global__ __launch_bounds__(256) void gat_aggregate(const int* __restrict__ src,
                                                     const int* __restrict__ dst,
                                                     const float* __restrict__ elog,
                                                     const float* __restrict__ den,
                                                     const float* __restrict__ Z,
                                                     float* __restrict__ acc,
                                                     int H, int hshift) {
  __shared__ float alpha[12];
  const int e = blockIdx.x;
  const int s = src[e], d = dst[e];
  if (threadIdx.x < (unsigned)H)
    alpha[threadIdx.x] = elog[e * H + threadIdx.x] / den[d * H + threadIdx.x];
  __syncthreads();
  for (int i = threadIdx.x; i < DIM; i += 256)
    atomicAdd(&acc[(size_t)d * DIM + i], alpha[i >> hshift] * Z[(size_t)s * DIM + i]);
}

// out = [elu](acc + z + bias); grid (NN, 3) x 256 threads = 768 cols
__global__ void gat_finalize(const float* __restrict__ acc, const float* __restrict__ Z,
                             const float* __restrict__ bias, float* __restrict__ out,
                             int use_elu) {
  const int n = blockIdx.x;
  const int c = blockIdx.y * 256 + threadIdx.x;
  size_t i = (size_t)n * DIM + c;
  float v = acc[i] + Z[i] + bias[c];
  if (use_elu) v = (v > 0.0f) ? v : expm1f(v);
  out[i] = v;
}

// ---------------------------------------------------------------------------
extern "C" void kernel_launch(void* const* d_in, const int* in_sizes, int n_in,
                              void* d_out, int out_size, void* d_ws, size_t ws_size,
                              hipStream_t stream) {
  (void)in_sizes; (void)n_in; (void)out_size; (void)ws_size;
  const float* x   = (const float*)d_in[0];
  const int*   src = (const int*)d_in[1];
  const int*   dst = (const int*)d_in[2];
  const float* Wl[3] = {(const float*)d_in[3], (const float*)d_in[7],  (const float*)d_in[11]};
  const float* al[3] = {(const float*)d_in[4], (const float*)d_in[8],  (const float*)d_in[12]};
  const float* ar[3] = {(const float*)d_in[5], (const float*)d_in[9],  (const float*)d_in[13]};
  const float* bl[3] = {(const float*)d_in[6], (const float*)d_in[10], (const float*)d_in[14]};
  float* out = (float*)d_out;

  char* ws = (char*)d_ws;
  float*  z    = (float*)ws;  ws += (size_t)NN * DIM * sizeof(float);
  float*  acc  = (float*)ws;  ws += (size_t)NN * DIM * sizeof(float);
  __bf16* xb   = (__bf16*)ws; ws += (size_t)NN * DIM * sizeof(__bf16);
  __bf16* wt   = (__bf16*)ws; ws += (size_t)DIM * DIM * sizeof(__bf16);
  float*  el   = (float*)ws;  ws += (size_t)NN * 12 * sizeof(float);
  float*  er   = (float*)ws;  ws += (size_t)NN * 12 * sizeof(float);
  float*  m    = (float*)ws;  ws += (size_t)NN * 12 * sizeof(float);
  float*  den  = (float*)ws;  ws += (size_t)NN * 12 * sizeof(float);
  float*  elog = (float*)ws;  ws += (size_t)NE * 12 * sizeof(float);

  const int Hs[3]     = {12, 12, 1};
  const int hshift[3] = {6, 6, 10};   // h = col >> hshift (D=64 -> 6; D=768,H=1 -> always 0)
  const float* layin = x;

  for (int L = 0; L < 3; ++L) {
    const int H = Hs[L];
    const int EH = NE * H;
    cvt_x_bf16 <<<(NN * DIM + 255) / 256, 256, 0, stream>>>(layin, xb, NN * DIM);
    cvt_w_t_bf16<<<(DIM * DIM + 255) / 256, 256, 0, stream>>>(Wl[L], wt);
    gat_gemm   <<<dim3(DIM / 128, (NN + 63) / 64), 256, 0, stream>>>(xb, wt, z);
    gat_zero   <<<(NN * DIM + 255) / 256, 256, 0, stream>>>(acc, NN * DIM);
    gat_attn   <<<NN, 256, 0, stream>>>(z, al[L], ar[L], el, er, m, den, H, hshift[L]);
    gat_edge_max<<<(EH + 255) / 256, 256, 0, stream>>>(src, dst, el, er, elog, m, H);
    gat_edge_exp<<<(EH + 255) / 256, 256, 0, stream>>>(dst, elog, m, den, H);
    gat_aggregate<<<NE, 256, 0, stream>>>(src, dst, elog, den, z, acc, H, hshift[L]);
    gat_finalize<<<dim3(NN, 3), 256, 0, stream>>>(acc, z, bl[L], out, (L < 2) ? 1 : 0);
    layin = out;   // d_out doubles as the activation ping buffer between layers
  }
}
